// GCN_graph3_6090263626391
// MI455X (gfx1250) — matrix operand order repaired and checked by reading, not compiled, not run on hardware
//
#include <hip/hip_runtime.h>
#include <hip/hip_bf16.h>
#include <cstdint>
#include <cstddef>

#define NN   20000
#define EE   320000
#define INC  256
#define HID  768
#define OUTC 7

typedef __attribute__((ext_vector_type(16))) _Float16 v16h;
typedef __attribute__((ext_vector_type(8)))  _Float16 v8h;
typedef __attribute__((ext_vector_type(8)))  float    v8f;

// ---------------- small utility kernels ----------------

__global__ void k_zero(float* p, int n) {
    int i = blockIdx.x * blockDim.x + threadIdx.x;
    if (i < n) p[i] = 0.0f;
}

__global__ void k_deg(const int* __restrict__ ei, float* __restrict__ deg, int E) {
    int i = blockIdx.x * blockDim.x + threadIdx.x;
    if (i < E) atomicAdd(&deg[ei[E + i]], 1.0f);   // dst side
}

__global__ void k_dis(const float* __restrict__ deg, float* __restrict__ dis, int n) {
    int i = blockIdx.x * blockDim.x + threadIdx.x;
    if (i < n) dis[i] = rsqrtf(deg[i] + 1.0f);     // A + I degrees
}

__global__ void k_f2h(const float* __restrict__ in, _Float16* __restrict__ out, int n) {
    int i = blockIdx.x * blockDim.x + threadIdx.x;
    if (i < n) out[i] = (_Float16)in[i];
}

// W is K x H row-major (f32); produce transposed f16: Wt[h*K + k]
__global__ void k_wt(const float* __restrict__ W, _Float16* __restrict__ Wt, int K, int H) {
    int idx = blockIdx.x * blockDim.x + threadIdx.x;
    if (idx >= K * H) return;
    int k = idx / H, h = idx - k * H;
    Wt[(size_t)h * K + k] = (_Float16)W[idx];
}

// ---------------- WMMA GEMM: C[nrows x H] = A[nrows x K] * B[K x H] ----------------
// A: f16 row-major. Bt: f16, H x K row-major (i.e. B transposed). C: f32 row-major.
// One wave computes a 16x64 strip: 4 accumulators of 16x16, K stepped by 32.
__global__ __launch_bounds__(256) void
k_gemm(const _Float16* __restrict__ A, const _Float16* __restrict__ Bt,
       float* __restrict__ C, int nrows, int K, int H) {
    const int lane = threadIdx.x & 31;
    const int wave = threadIdx.x >> 5;
    const int rowTile = blockIdx.y * 8 + wave;
    if (rowTile * 16 >= nrows) return;           // uniform per wave: EXEC stays all-ones
    const int rowBase = rowTile * 16;
    const int colBase = blockIdx.x * 64;

    const int m   = lane & 15;                   // A row / B column / C column within tile
    const int hiA = (lane >> 4) * 16;            // A K-group offset (0 or 16)
    const int hiB = (lane >> 4) * 16;            // B K-group offset (0 or 16)

    v8f acc[4];
#pragma unroll
    for (int t = 0; t < 4; ++t) acc[t] = (v8f){};

    const _Float16* arow = A + (size_t)(rowBase + m) * K;

    for (int k0 = 0; k0 < K; k0 += 32) {
        // A fragment: halves 0..15 -> K = k0+hiA .. +15 (16 consecutive K values)
        v16h a;
        {
            const v8h a0 = *(const v8h*)(arow + k0 + hiA);
            const v8h a1 = *(const v8h*)(arow + k0 + hiA + 8);
#pragma unroll
            for (int i = 0; i < 8; ++i) { a[i] = a0[i]; a[8 + i] = a1[i]; }
        }
#pragma unroll
        for (int t = 0; t < 4; ++t) {
            // B fragment for column (colBase + t*16 + m): 16 consecutive K values
            const _Float16* bp = Bt + (size_t)(colBase + t * 16 + m) * K + k0 + hiB;
            v16h b;
            const v8h b0 = *(const v8h*)(bp);
            const v8h b1 = *(const v8h*)(bp + 8);
#pragma unroll
            for (int i = 0; i < 8; ++i) { b[i] = b0[i]; b[8 + i] = b1[i]; }
            acc[t] = __builtin_amdgcn_wmma_f32_16x16x32_f16(
                false, a, false, b, (short)0, acc[t], false, false);
        }
    }

    // C layout: lane n = lane&15 is column; VGPR r is row r (+8 for lanes 16..31)
    const int rOff = (lane >> 4) * 8;
#pragma unroll
    for (int t = 0; t < 4; ++t) {
#pragma unroll
        for (int r = 0; r < 8; ++r) {
            C[(size_t)(rowBase + r + rOff) * H + colBase + t * 16 + m] = acc[t][r];
        }
    }
}

// ---------------- aggregation ----------------

// self-loop term: Agg[i][c] = H[i][c] * dis[i]^2   (also serves as the zero-init)
__global__ void k_self(const float* __restrict__ Hb, const float* __restrict__ dis,
                       float* __restrict__ Ag, int nrows, int Hdim) {
    int idx = blockIdx.x * blockDim.x + threadIdx.x;
    if (idx >= nrows * Hdim) return;
    int row = idx / Hdim;
    float d = dis[row];
    Ag[idx] = Hb[idx] * d * d;
}

// one wave per edge: Ag[dst][c] += H[src][c] * dis[src]*dis[dst]
__global__ __launch_bounds__(256) void
k_edge(const int* __restrict__ ei, const float* __restrict__ Hb,
       const float* __restrict__ dis, float* __restrict__ Ag, int E, int Hdim) {
    int e = blockIdx.x * 8 + (threadIdx.x >> 5);
    if (e >= E) return;
    int lane = threadIdx.x & 31;
    int s = ei[e], d = ei[E + e];
    float nrm = dis[s] * dis[d];
    const float* hs = Hb + (size_t)s * Hdim;
    float* ad = Ag + (size_t)d * Hdim;
    for (int c = lane; c < Hdim; c += 32)
        atomicAdd(&ad[c], hs[c] * nrm);
}

// ---------------- LayerNorm + ReLU, f16 output (wave per row) ----------------
__global__ __launch_bounds__(256) void
k_ln(const float* __restrict__ Ag, const float* __restrict__ bias,
     const float* __restrict__ g, const float* __restrict__ be,
     _Float16* __restrict__ Y, int nrows) {
    int row = blockIdx.x * 8 + (threadIdx.x >> 5);
    if (row >= nrows) return;
    int lane = threadIdx.x & 31;
    const float* x = Ag + (size_t)row * HID;

    float v[HID / 32];
    float s = 0.0f;
#pragma unroll
    for (int i = 0; i < HID / 32; ++i) {
        int c = lane + 32 * i;
        v[i] = x[c] + bias[c];
        s += v[i];
    }
#pragma unroll
    for (int o = 16; o > 0; o >>= 1) s += __shfl_xor(s, o, 32);
    float mu = s * (1.0f / HID);

    float vs = 0.0f;
#pragma unroll
    for (int i = 0; i < HID / 32; ++i) { float dd = v[i] - mu; vs += dd * dd; }
#pragma unroll
    for (int o = 16; o > 0; o >>= 1) vs += __shfl_xor(vs, o, 32);
    float inv = rsqrtf(vs * (1.0f / HID) + 1e-5f);

    _Float16* y = Y + (size_t)row * HID;
#pragma unroll
    for (int i = 0; i < HID / 32; ++i) {
        int c = lane + 32 * i;
        float o = (v[i] - mu) * inv * g[c] + be[c];
        y[c] = (_Float16)fmaxf(o, 0.0f);
    }
}

// ---------------- pooling + final linear ----------------

__global__ __launch_bounds__(256) void
k_pool(const _Float16* __restrict__ Y, float* __restrict__ pooled, int nrows) {
    int nodeBase = blockIdx.x * 32;
    int t = threadIdx.x;
    for (int c = t; c < HID; c += 256) {
        float s = 0.0f;
        for (int r = 0; r < 32; ++r) {
            int row = nodeBase + r;
            if (row < nrows) s += (float)Y[(size_t)row * HID + c];
        }
        atomicAdd(&pooled[c], s);
    }
}

__global__ __launch_bounds__(256) void
k_final(const float* __restrict__ pooled, const float* __restrict__ Wl,
        const float* __restrict__ bl, float* __restrict__ out, float invN) {
    __shared__ float red[256];
    int t = threadIdx.x;
    for (int o = 0; o < OUTC; ++o) {
        float s = 0.0f;
        for (int c = t; c < HID; c += 256)
            s += pooled[c] * invN * Wl[c * OUTC + o];
        red[t] = s;
        __syncthreads();
        for (int k = 128; k > 0; k >>= 1) {
            if (t < k) red[t] += red[t + k];
            __syncthreads();
        }
        if (t == 0) out[o] = red[0] + bl[o];
        __syncthreads();
    }
}

// ---------------- launch ----------------

static inline char* align_up(char* p, size_t a) {
    return (char*)(((uintptr_t)p + a - 1) & ~(uintptr_t)(a - 1));
}

extern "C" void kernel_launch(void* const* d_in, const int* in_sizes, int n_in,
                              void* d_out, int out_size, void* d_ws, size_t ws_size,
                              hipStream_t stream) {
    const float* x   = (const float*)d_in[0];
    const int*   ei  = (const int*)d_in[1];
    const float* W1  = (const float*)d_in[2];
    const float* b1  = (const float*)d_in[3];
    const float* W2  = (const float*)d_in[4];
    const float* b2  = (const float*)d_in[5];
    const float* W3  = (const float*)d_in[6];
    const float* b3  = (const float*)d_in[7];
    const float* g1  = (const float*)d_in[8];
    const float* be1 = (const float*)d_in[9];
    const float* g2  = (const float*)d_in[10];
    const float* be2 = (const float*)d_in[11];
    const float* g3  = (const float*)d_in[12];
    const float* be3 = (const float*)d_in[13];
    const float* Wl  = (const float*)d_in[14];
    const float* bl  = (const float*)d_in[15];
    float* out = (float*)d_out;

    char* p = (char*)d_ws;
    auto carve = [&](size_t bytes) -> void* {
        p = align_up(p, 256);
        void* r = (void*)p;
        p += bytes;
        return r;
    };
    float*    deg    = (float*)carve((size_t)NN * 4);
    float*    dis    = (float*)carve((size_t)NN * 4);
    float*    pooled = (float*)carve((size_t)HID * 4);
    _Float16* X16    = (_Float16*)carve((size_t)NN * INC * 2);
    _Float16* Wt1    = (_Float16*)carve((size_t)INC * HID * 2);
    _Float16* Wt2    = (_Float16*)carve((size_t)HID * HID * 2);
    _Float16* Wt3    = (_Float16*)carve((size_t)HID * HID * 2);
    float*    Hb     = (float*)carve((size_t)NN * HID * 4);
    float*    Ab     = (float*)carve((size_t)NN * HID * 4);
    _Float16* Y16    = (_Float16*)carve((size_t)NN * HID * 2);

    // ---- degrees / norms ----
    k_zero<<<(NN + 255) / 256, 256, 0, stream>>>(deg, NN);
    k_zero<<<(HID + 255) / 256, 256, 0, stream>>>(pooled, HID);
    k_deg<<<(EE + 255) / 256, 256, 0, stream>>>(ei, deg, EE);
    k_dis<<<(NN + 255) / 256, 256, 0, stream>>>(deg, dis, NN);

    // ---- precision conversions ----
    k_f2h<<<((size_t)NN * INC + 255) / 256, 256, 0, stream>>>(x, X16, NN * INC);
    k_wt<<<((size_t)INC * HID + 255) / 256, 256, 0, stream>>>(W1, Wt1, INC, HID);
    k_wt<<<((size_t)HID * HID + 255) / 256, 256, 0, stream>>>(W2, Wt2, HID, HID);
    k_wt<<<((size_t)HID * HID + 255) / 256, 256, 0, stream>>>(W3, Wt3, HID, HID);

    const dim3 gemmBlk(256, 1, 1);
    const dim3 gemmGrd(HID / 64, (NN / 16 + 7) / 8, 1);
    const int selfBlocks = (int)(((size_t)NN * HID + 255) / 256);
    const int edgeBlocks = (EE + 7) / 8;
    const int lnBlocks   = (NN + 7) / 8;

    // ---- layer 1 ----
    k_gemm<<<gemmGrd, gemmBlk, 0, stream>>>(X16, Wt1, Hb, NN, INC, HID);
    k_self<<<selfBlocks, 256, 0, stream>>>(Hb, dis, Ab, NN, HID);
    k_edge<<<edgeBlocks, 256, 0, stream>>>(ei, Hb, dis, Ab, EE, HID);
    k_ln<<<lnBlocks, 256, 0, stream>>>(Ab, b1, g1, be1, Y16, NN);

    // ---- layer 2 ----
    k_gemm<<<gemmGrd, gemmBlk, 0, stream>>>(Y16, Wt2, Hb, NN, HID, HID);
    k_self<<<selfBlocks, 256, 0, stream>>>(Hb, dis, Ab, NN, HID);
    k_edge<<<edgeBlocks, 256, 0, stream>>>(ei, Hb, dis, Ab, EE, HID);
    k_ln<<<lnBlocks, 256, 0, stream>>>(Ab, b2, g2, be2, Y16, NN);

    // ---- layer 3 ----
    k_gemm<<<gemmGrd, gemmBlk, 0, stream>>>(Y16, Wt3, Hb, NN, HID, HID);
    k_self<<<selfBlocks, 256, 0, stream>>>(Hb, dis, Ab, NN, HID);
    k_edge<<<edgeBlocks, 256, 0, stream>>>(ei, Hb, dis, Ab, EE, HID);
    k_ln<<<lnBlocks, 256, 0, stream>>>(Ab, b3, g3, be3, Y16, NN);

    // ---- pool + classify ----
    k_pool<<<(NN + 31) / 32, 256, 0, stream>>>(Y16, pooled, NN);
    k_final<<<1, 256, 0, stream>>>(pooled, Wl, bl, out, 1.0f / (float)NN);
}
